// ContextAdaptiveGating_69569880261161
// MI455X (gfx1250) — compile-verified, matrix operand order checked by reading, not run on hardware
//
#include <hip/hip_runtime.h>
#include <hip/hip_bf16.h>
#include <math.h>

typedef __attribute__((ext_vector_type(16))) __bf16 v16bf;
typedef __attribute__((ext_vector_type(8)))  float  v8f;

#define HEADS    16
#define HD       128
#define GATE_IN  2608
#define KPAD     2624   // 82 * 32
#define GATE_HID 1024
#define OUT_DIM  80
#define NTOK     8192   // B*L = 2*4096

// ---------------------------------------------------------------------------
// helpers
// ---------------------------------------------------------------------------
__device__ __forceinline__ void bf16_split(float x, __bf16& hi, __bf16& lo) {
  __bf16 h = (__bf16)x;          // RN float->bf16
  hi = h;
  lo = (__bf16)(x - (float)h);   // residual -> second bf16
}

__device__ __forceinline__ float wave_sum(float v) {
#pragma unroll
  for (int off = 16; off > 0; off >>= 1) v += __shfl_xor(v, off, 32);
  return v;
}

// ---------------------------------------------------------------------------
// Kernel 1: f32 -> (bf16 hi, bf16 lo) with K padding (weights)
// ---------------------------------------------------------------------------
__global__ void __launch_bounds__(256) split_pad_kernel(
    const float* __restrict__ src,
    __bf16* __restrict__ hi, __bf16* __restrict__ lo,
    int rows, int srcK, int dstK) {
  long long idx = (long long)blockIdx.x * blockDim.x + threadIdx.x;
  long long total = (long long)rows * dstK;
  if (idx >= total) return;
  int k = (int)(idx % dstK);
  long long r = idx / dstK;
  float x = (k < srcK) ? src[r * srcK + k] : 0.0f;
  __bf16 h, l;
  bf16_split(x, h, l);
  hi[idx] = h;
  lo[idx] = l;
}

// ---------------------------------------------------------------------------
// Kernel 2: build gate-input matrix G[NTOK, KPAD] as bf16 hi/lo
//   layout per token: [0,2048) hidden | 2048+h*20+p*4+s stats |
//                     2368+h*5+p l2 | 2448+h*10+j pairwise | [2608,2624)=0
// one block (256 threads = 8 waves) per token; each wave handles 2 heads
// ---------------------------------------------------------------------------
__global__ void __launch_bounds__(256) features_kernel(
    const float* __restrict__ hs,
    const float* __restrict__ p0, const float* __restrict__ p1,
    const float* __restrict__ p2, const float* __restrict__ p3,
    const float* __restrict__ p4,
    __bf16* __restrict__ Ghi, __bf16* __restrict__ Glo) {
  const int tok = blockIdx.x;
  const int tid = threadIdx.x;
  const long long gbase = (long long)tok * KPAD;

  // hidden part: straight split-convert
  for (int i = tid; i < 2048; i += 256) {
    __bf16 h, l;
    bf16_split(hs[(long long)tok * 2048 + i], h, l);
    Ghi[gbase + i] = h;
    Glo[gbase + i] = l;
  }
  // zero pad 2608..2623
  if (tid < (KPAD - GATE_IN)) {
    Ghi[gbase + GATE_IN + tid] = (__bf16)0.0f;
    Glo[gbase + GATE_IN + tid] = (__bf16)0.0f;
  }

  const int wave = tid >> 5, lane = tid & 31;
  const float* P[5] = {p0, p1, p2, p3, p4};

#pragma unroll
  for (int hh = 0; hh < 2; ++hh) {
    const int h = wave * 2 + hh;
    const long long base = ((long long)tok * HEADS + h) * HD;

    float x[5][4];
#pragma unroll
    for (int p = 0; p < 5; ++p)
#pragma unroll
      for (int j = 0; j < 4; ++j) x[p][j] = P[p][base + lane + 32 * j];

    float s[5], ss[5], sa[5], dot[10];
#pragma unroll
    for (int p = 0; p < 5; ++p) {
      float a = 0.f, b = 0.f, c = 0.f;
#pragma unroll
      for (int j = 0; j < 4; ++j) {
        float v = x[p][j];
        a += v; b += v * v; c += fabsf(v);
      }
      s[p] = wave_sum(a);
      ss[p] = wave_sum(b);
      sa[p] = wave_sum(c);
    }
    {
      int jj = 0;
#pragma unroll
      for (int p = 0; p < 5; ++p)
#pragma unroll
        for (int q = p + 1; q < 5; ++q) {
          float d = 0.f;
#pragma unroll
          for (int j = 0; j < 4; ++j) d += x[p][j] * x[q][j];
          dot[jj] = wave_sum(d);
          ++jj;
        }
    }

    if (lane == 0) {
      __bf16 bh, bl;
#pragma unroll
      for (int p = 0; p < 5; ++p) {
        float mean = s[p] * (1.0f / 128.0f);
        float var  = ss[p] * (1.0f / 128.0f) - mean * mean;  // unbiased=False
        float am   = sa[p] * (1.0f / 128.0f);
        float l2   = sqrtf(ss[p]);
        long long o = gbase + 2048 + h * 20 + p * 4;
        bf16_split(mean, bh, bl); Ghi[o + 0] = bh; Glo[o + 0] = bl;
        bf16_split(var,  bh, bl); Ghi[o + 1] = bh; Glo[o + 1] = bl;
        bf16_split(am,   bh, bl); Ghi[o + 2] = bh; Glo[o + 2] = bl;
        bf16_split(l2,   bh, bl); Ghi[o + 3] = bh; Glo[o + 3] = bl;
        long long on = gbase + 2368 + h * 5 + p;
        bf16_split(l2,   bh, bl); Ghi[on] = bh; Glo[on] = bl;
      }
      int jj = 0;
#pragma unroll
      for (int p = 0; p < 5; ++p)
#pragma unroll
        for (int q = p + 1; q < 5; ++q) {
          float d2 = ss[p] + ss[q] - 2.0f * dot[jj];
          float dist = sqrtf(fmaxf(d2, 0.0f));
          long long o = gbase + 2448 + h * 10 + jj;
          bf16_split(dist, bh, bl); Ghi[o] = bh; Glo[o] = bl;
          ++jj;
        }
    }
  }
}

// ---------------------------------------------------------------------------
// bf16x3 GEMM:  C[M,N] = A[M,K] * B[N,K]^T  (+bias, epilogue)
//   A,B stored as bf16 hi/lo row-major (K contiguous, even ld)
//   fragment layout per CDNA5 ISA (16-bit A 16x32): lane = row(M/N),
//   K = (lane>=16 ? 8 : 0) + (v&3)*2 + (v>=4 ? 16 : 0)  -> v0..3 / v4..7
//   coalesce to two b128 loads per fragment row
//   EPI==0: exact GELU(x+bias) -> bf16 hi/lo;   EPI==1: f32 (x+bias)
//   B fragments loaded just-in-time to keep live ranges short (no spills);
//   __launch_bounds__(128,1) lets the backend use the full VGPR file.
// ---------------------------------------------------------------------------
__device__ __forceinline__ void load_frag_u(const __bf16* __restrict__ rowptr,
                                            int kbase, unsigned int u[8]) {
  const unsigned int* p = (const unsigned int*)rowptr;
#pragma unroll
  for (int v = 0; v < 8; ++v) {
    int k = kbase + (v & 3) * 2 + ((v & 4) ? 16 : 0);
    u[v] = p[k >> 1];
  }
}

template <int TM, int TN, int EPI>
__global__ void __launch_bounds__(128, 1) gemm_bf16x3_kernel(
    const __bf16* __restrict__ Ahi, const __bf16* __restrict__ Alo,
    const __bf16* __restrict__ Bhi, const __bf16* __restrict__ Blo,
    const float* __restrict__ bias,
    __bf16* __restrict__ Ohi, __bf16* __restrict__ Olo,
    float* __restrict__ Of,
    int M, int N, int K, int lda, int ldb, int ldc) {
  const int lane  = threadIdx.x & 31;
  const int gwave = (int)((blockIdx.x * blockDim.x + threadIdx.x) >> 5);
  const int mtiles = M / (16 * TM);
  const int ntiles = (N + 16 * TN - 1) / (16 * TN);
  const int nt = gwave % ntiles;
  const int mt = gwave / ntiles;
  if (mt >= mtiles) return;   // wave-uniform: EXEC stays all-1s for WMMA
  const int m0 = mt * 16 * TM;
  const int n0 = nt * 16 * TN;
  const int row = lane & 15;
  const int khalf = (lane >> 4) * 8;

  const __bf16* arow_hi[TM];
  const __bf16* arow_lo[TM];
#pragma unroll
  for (int i = 0; i < TM; ++i) {
    long long r = m0 + i * 16 + row;
    arow_hi[i] = Ahi + r * lda;
    arow_lo[i] = Alo + r * lda;
  }
  const __bf16* brow_hi[TN];
  const __bf16* brow_lo[TN];
#pragma unroll
  for (int j = 0; j < TN; ++j) {
    long long r = n0 + j * 16 + row;   // all n-rows in range by construction
    brow_hi[j] = Bhi + r * ldb;
    brow_lo[j] = Blo + r * ldb;
  }

  v8f acc[TM][TN];
#pragma unroll
  for (int i = 0; i < TM; ++i)
#pragma unroll
    for (int j = 0; j < TN; ++j)
#pragma unroll
      for (int e = 0; e < 8; ++e) acc[i][j][e] = 0.0f;

  union Frag { unsigned int u[8]; v16bf v; };

  for (int kk = 0; kk < K; kk += 32) {
    const int kbase = kk + khalf;
    Frag ah[TM], al[TM];
#pragma unroll
    for (int i = 0; i < TM; ++i) {
      load_frag_u(arow_hi[i], kbase, ah[i].u);
      load_frag_u(arow_lo[i], kbase, al[i].u);
    }
#pragma unroll
    for (int j = 0; j < TN; ++j) {
      Frag bh, bl;   // short-lived: loaded and consumed inside this j-step
      load_frag_u(brow_hi[j], kbase, bh.u);
      load_frag_u(brow_lo[j], kbase, bl.u);
#pragma unroll
      for (int i = 0; i < TM; ++i) {
        acc[i][j] = __builtin_amdgcn_wmma_f32_16x16x32_bf16(
            false, ah[i].v, false, bh.v, (short)0, acc[i][j], false, false);
        acc[i][j] = __builtin_amdgcn_wmma_f32_16x16x32_bf16(
            false, al[i].v, false, bh.v, (short)0, acc[i][j], false, false);
        acc[i][j] = __builtin_amdgcn_wmma_f32_16x16x32_bf16(
            false, ah[i].v, false, bl.v, (short)0, acc[i][j], false, false);
      }
    }
  }

  // epilogue: C layout -> lane holds n = lane&15, VGPR r holds m = r + 8*(lane>>4)
#pragma unroll
  for (int j = 0; j < TN; ++j) {
    int n = n0 + j * 16 + (lane & 15);
    if (n >= N) continue;
    float bv = bias[n];
#pragma unroll
    for (int i = 0; i < TM; ++i) {
#pragma unroll
      for (int r = 0; r < 8; ++r) {
        int m = m0 + i * 16 + r + 8 * (lane >> 4);
        float x = acc[i][j][r] + bv;
        long long o = (long long)m * ldc + n;
        if (EPI == 0) {
          float g = 0.5f * x * (1.0f + erff(x * 0.70710678118654752f));  // exact GELU
          __bf16 h, l;
          bf16_split(g, h, l);
          Ohi[o] = h;
          Olo[o] = l;
        } else {
          Of[o] = x;
        }
      }
    }
  }
}

// ---------------------------------------------------------------------------
// Kernel 4: softmax(temp)/clip/renorm + weighted path combine; reg_loss = 0
// ---------------------------------------------------------------------------
__global__ void __launch_bounds__(256) finalize_kernel(
    const float* __restrict__ logits,
    const float* __restrict__ p0, const float* __restrict__ p1,
    const float* __restrict__ p2, const float* __restrict__ p3,
    const float* __restrict__ p4,
    const float* __restrict__ log_temp,
    float* __restrict__ out) {
  const int tok = blockIdx.x;
  const int wave = threadIdx.x >> 5, lane = threadIdx.x & 31;
  const float* P[5] = {p0, p1, p2, p3, p4};

#pragma unroll
  for (int hh = 0; hh < 2; ++hh) {
    const int h = wave * 2 + hh;
    const float invt = expf(-log_temp[h]);   // 1/exp(log_temp)
    const float* lg = logits + (long long)tok * OUT_DIM + h * 5;
    float z[5];
#pragma unroll
    for (int p = 0; p < 5; ++p) z[p] = lg[p] * invt;
    float mx = z[0];
#pragma unroll
    for (int p = 1; p < 5; ++p) mx = fmaxf(mx, z[p]);
    float e[5], se = 0.f;
#pragma unroll
    for (int p = 0; p < 5; ++p) { e[p] = expf(z[p] - mx); se += e[p]; }
    float pr[5], s2 = 0.f;
    float inv = 1.0f / se;
#pragma unroll
    for (int p = 0; p < 5; ++p) { pr[p] = fmaxf(e[p] * inv, 0.02f); s2 += pr[p]; }
    float inv2 = 1.0f / s2;
#pragma unroll
    for (int p = 0; p < 5; ++p) pr[p] *= inv2;

    const long long base = ((long long)tok * HEADS + h) * HD;
#pragma unroll
    for (int j = 0; j < 4; ++j) {
      int d = lane + 32 * j;
      float o = 0.f;
#pragma unroll
      for (int p = 0; p < 5; ++p) o += pr[p] * P[p][base + d];
      out[base + d] = o;
    }
  }
  if (tok == 0 && threadIdx.x == 0) out[(long long)NTOK * HEADS * HD] = 0.0f;  // reg_loss
}

// ---------------------------------------------------------------------------
// launch
// ---------------------------------------------------------------------------
extern "C" void kernel_launch(void* const* d_in, const int* in_sizes, int n_in,
                              void* d_out, int out_size, void* d_ws, size_t ws_size,
                              hipStream_t stream) {
  (void)in_sizes; (void)n_in; (void)out_size; (void)ws_size;
  const float* hs = (const float*)d_in[0];
  const float* p0 = (const float*)d_in[1];
  const float* p1 = (const float*)d_in[2];
  const float* p2 = (const float*)d_in[3];
  const float* p3 = (const float*)d_in[4];
  const float* p4 = (const float*)d_in[5];
  const float* W1 = (const float*)d_in[6];
  const float* b1 = (const float*)d_in[7];
  const float* W2 = (const float*)d_in[8];
  const float* b2 = (const float*)d_in[9];
  const float* log_temp = (const float*)d_in[10];
  float* out = (float*)d_out;

  char* ws = (char*)d_ws;
  size_t off = 0;
  auto take = [&](size_t bytes) -> char* {
    off = (off + 255) & ~(size_t)255;
    char* p = ws + off;
    off += bytes;
    return p;
  };
  __bf16* Ghi  = (__bf16*)take((size_t)NTOK * KPAD * 2);
  __bf16* Glo  = (__bf16*)take((size_t)NTOK * KPAD * 2);
  __bf16* W1hi = (__bf16*)take((size_t)GATE_HID * KPAD * 2);
  __bf16* W1lo = (__bf16*)take((size_t)GATE_HID * KPAD * 2);
  __bf16* W2hi = (__bf16*)take((size_t)OUT_DIM * GATE_HID * 2);
  __bf16* W2lo = (__bf16*)take((size_t)OUT_DIM * GATE_HID * 2);
  __bf16* Hhi  = (__bf16*)take((size_t)NTOK * GATE_HID * 2);
  __bf16* Hlo  = (__bf16*)take((size_t)NTOK * GATE_HID * 2);
  float*  lgts = (float*)take((size_t)NTOK * OUT_DIM * 4);

  // weight prep
  {
    long long tot = (long long)GATE_HID * KPAD;
    split_pad_kernel<<<(unsigned)((tot + 255) / 256), 256, 0, stream>>>(
        W1, W1hi, W1lo, GATE_HID, GATE_IN, KPAD);
  }
  {
    long long tot = (long long)OUT_DIM * GATE_HID;
    split_pad_kernel<<<(unsigned)((tot + 255) / 256), 256, 0, stream>>>(
        W2, W2hi, W2lo, OUT_DIM, GATE_HID, GATE_HID);
  }
  // features
  features_kernel<<<NTOK, 256, 0, stream>>>(hs, p0, p1, p2, p3, p4, Ghi, Glo);
  // GEMM1: [8192 x 2624] x [2624 -> 1024], 32x32 wave tiles, GELU epilogue
  {
    int mtiles = NTOK / 32, ntiles = GATE_HID / 32;
    int threads = mtiles * ntiles * 32;
    gemm_bf16x3_kernel<2, 2, 0><<<(threads + 127) / 128, 128, 0, stream>>>(
        Ghi, Glo, W1hi, W1lo, b1, Hhi, Hlo, nullptr,
        NTOK, GATE_HID, KPAD, KPAD, KPAD, GATE_HID);
  }
  // GEMM2: [8192 x 1024] x [1024 -> 80], 32x16 wave tiles, f32 logits epilogue
  {
    int mtiles = NTOK / 32, ntiles = OUT_DIM / 16;
    int threads = mtiles * ntiles * 32;
    gemm_bf16x3_kernel<2, 1, 1><<<(threads + 127) / 128, 128, 0, stream>>>(
        Hhi, Hlo, W2hi, W2lo, b2, nullptr, nullptr, lgts,
        NTOK, OUT_DIM, GATE_HID, GATE_HID, GATE_HID, OUT_DIM);
  }
  // softmax + combine
  finalize_kernel<<<NTOK, 256, 0, stream>>>(lgts, p0, p1, p2, p3, p4, log_temp, out);
}
